// ESNPCA_39204461478205
// MI455X (gfx1250) — matrix operand order repaired and checked by reading, not compiled, not run on hardware
//
#include <hip/hip_runtime.h>
#include <math.h>

#define TT   4096   // time steps
#define RDIM 4096   // reservoir size
#define NIN  3      // input size
#define OUTW (RDIM + NIN)

typedef __attribute__((ext_vector_type(16))) __bf16  v16bf;
typedef __attribute__((ext_vector_type(8)))  float   v8f;
typedef __attribute__((ext_vector_type(4)))  unsigned int u32x4;

__device__ __forceinline__ unsigned int f32_to_bf16_rne(float f) {
    unsigned int u = __builtin_bit_cast(unsigned int, f);
    unsigned int r = u + 0x7FFFu + ((u >> 16) & 1u);
    return r >> 16;
}

// One-time setup: pack Wh to bf16 pairs, zero h0, copy x into the tail columns of out.
__global__ void esn_init(const float* __restrict__ Wh,
                         const float* __restrict__ x,
                         unsigned int* __restrict__ wh_pk,
                         float* __restrict__ h0,
                         float* __restrict__ out) {
    size_t id     = (size_t)blockIdx.x * blockDim.x + threadIdx.x;
    size_t stride = (size_t)gridDim.x * blockDim.x;

    const size_t NPK = (size_t)RDIM * RDIM / 2;
    for (size_t j = id; j < NPK; j += stride) {
        unsigned int lo = f32_to_bf16_rne(Wh[2 * j]);
        unsigned int hi = f32_to_bf16_rne(Wh[2 * j + 1]);
        wh_pk[j] = lo | (hi << 16);
    }
    for (size_t j = id; j < RDIM; j += stride) h0[j] = 0.0f;
    for (size_t j = id; j < (size_t)TT * NIN; j += stride) {
        size_t t = j / NIN, i = j % NIN;
        out[t * (size_t)OUTW + RDIM + i] = x[j];
    }
}

// One recurrence step: h_out = LEAK*h_in + (1-LEAK)*tanh(Win@x_t + Wh@h_in)
// 4 waves / block, 16 rows / wave, 64 blocks -> 4096 rows.
__global__ __launch_bounds__(128) void esn_step(
        const unsigned short* __restrict__ whb,   // Wh as bf16, row-major [RDIM x RDIM]
        const float* __restrict__ hin,
        float* __restrict__ hout,
        const float* __restrict__ x,
        const float* __restrict__ Win,
        float* __restrict__ out,
        int t) {
    __shared__ unsigned int hpk[RDIM / 2];        // h_in packed as bf16 pairs (8 KB)

    for (int j = threadIdx.x; j < RDIM / 2; j += 128) {
        unsigned int lo = f32_to_bf16_rne(hin[2 * j]);
        unsigned int hi = f32_to_bf16_rne(hin[2 * j + 1]);
        hpk[j] = lo | (hi << 16);
    }
    __syncthreads();

    const int wave = threadIdx.x >> 5;
    const int lane = threadIdx.x & 31;
    const int tile = blockIdx.x * 4 + wave;       // 0..255
    const int row0 = tile * 16;

    // A layout (16-bit, 16x32): lane m<16 -> row m, K {k0..k0+7, k0+16..k0+23};
    //                           lane m+16 -> row m, K {k0+8..k0+15, k0+24..k0+31}.
    const int arow = row0 + (lane & 15);
    const int aoff = (lane < 16) ? 0 : 8;
    // B layout (32x16 bf16): lanes 0-15 hold K {0..7, 16..23}, lanes 16-31 hold K {8..15, 24..31} (2/VGPR).
    const int boff = (lane < 16) ? 0 : 8;
    const unsigned short* aptr = whb + (size_t)arow * RDIM + aoff;

    v8f c = {};
#pragma unroll 4
    for (int k0 = 0; k0 < RDIM; k0 += 32) {
        union { v16bf v; u32x4 q[2]; } A, B;
        A.q[0] = *(const u32x4*)(aptr + k0);
        A.q[1] = *(const u32x4*)(aptr + k0 + 16);
        const u32x4* bp = (const u32x4*)(&hpk[(k0 + boff) >> 1]);
        B.q[0] = bp[0];
        B.q[1] = bp[2];
        // D = A x B + C ; every B column replicates the h segment, so every
        // D column holds the 16 partial row-sums.
        c = __builtin_amdgcn_wmma_f32_16x16x32_bf16(
                /*neg_a=*/false, A.v, /*neg_b=*/false, B.v,
                /*c_mod=*/(short)0, c, /*reuse_a=*/false, /*reuse_b=*/false);
    }

    // D layout: VGPR j, lanes 0-15 -> M=j ; lanes 16-31 -> M=8+j.
    if (lane == 0 || lane == 16) {
        const int rbase = row0 + ((lane == 0) ? 0 : 8);
        const float x0 = x[(size_t)t * NIN + 0];
        const float x1 = x[(size_t)t * NIN + 1];
        const float x2 = x[(size_t)t * NIN + 2];
#pragma unroll
        for (int j = 0; j < 8; ++j) {
            const int r = rbase + j;
            const float u  = Win[r * NIN + 0] * x0 + Win[r * NIN + 1] * x1 +
                             Win[r * NIN + 2] * x2;
            const float hn = tanhf(u + c[j]);
            const float hv = 0.1f * hin[r] + 0.9f * hn;
            hout[r] = hv;
            out[(size_t)t * OUTW + r] = hv;
        }
    }
}

extern "C" void kernel_launch(void* const* d_in, const int* in_sizes, int n_in,
                              void* d_out, int out_size, void* d_ws, size_t ws_size,
                              hipStream_t stream) {
    const float* x   = (const float*)d_in[0];   // [T, 3]
    const float* Win = (const float*)d_in[1];   // [R, 3]
    const float* Wh  = (const float*)d_in[2];   // [R, R]
    float* out = (float*)d_out;                 // [T, R+3]

    unsigned int* wh_pk = (unsigned int*)d_ws;                      // 32 MB
    float* h0 = (float*)((char*)d_ws + (size_t)RDIM * RDIM * 2);    // 16 KB
    float* h1 = h0 + RDIM;                                          // 16 KB

    esn_init<<<4096, 256, 0, stream>>>(Wh, x, wh_pk, h0, out);

    const unsigned short* whb = (const unsigned short*)wh_pk;
    for (int t = 0; t < TT; ++t) {
        const float* hi = (t & 1) ? h1 : h0;
        float*       ho = (t & 1) ? h0 : h1;
        esn_step<<<64, 128, 0, stream>>>(whb, hi, ho, x, Win, out, t);
    }
}
